// RelationNetworkBaseline_54915451846824
// MI455X (gfx1250) — matrix operand order repaired and checked by reading, not compiled, not run on hardware
//
#include <hip/hip_runtime.h>

// ---------------------------------------------------------------------------
// Types for CDNA5 WMMA (wave32): v_wmma_f32_16x16x32_bf16
// ---------------------------------------------------------------------------
typedef __attribute__((ext_vector_type(16))) __bf16 v16bf;
typedef __attribute__((ext_vector_type(8)))  float  v8f;
typedef __attribute__((ext_vector_type(4)))  int    v4i;

union Frag {
    v16bf bf;
    v4i   i4[2];   // 32 bytes = two b128 loads
};

__device__ __forceinline__ unsigned short f2bf(float f) {
    union { float f; unsigned u; } v; v.f = f;
    unsigned r = v.u + 0x7FFFu + ((v.u >> 16) & 1u);   // round-to-nearest-even
    return (unsigned short)(r >> 16);
}

__device__ __forceinline__ v8f wmma_bf16(v16bf a, v16bf b, v8f c) {
    // (neg_a, A, neg_b, B, c_mod, C, reuse_a, reuse_b)
    return __builtin_amdgcn_wmma_f32_16x16x32_bf16(false, a, false, b,
                                                   (short)0, c, false, false);
}

#define VZERO8 {0.f,0.f,0.f,0.f,0.f,0.f,0.f,0.f}

// ---------------------------------------------------------------------------
// Problem constants
// ---------------------------------------------------------------------------
#define BB    32
#define NN    25
#define DD    512
#define HH    512
#define CC    1000
#define CPAD  1008
#define ROWS  (BB * NN)          // 800
#define PAIRS (NN * NN)          // 625
#define MTILE 64                 // pairs per workgroup in fused kernel
#define PT64  10                 // ceil(625/64)

// ---------------------------------------------------------------------------
// Prep: transpose-convert f32 weight [K+kOff][ld] -> bf16 Wt[n][k] (n<Npad)
// ---------------------------------------------------------------------------
__global__ void k_transpose_bf16(unsigned short* __restrict__ dst,
                                 const float* __restrict__ src,
                                 int kOff, int ld, int Npad, int Nvalid, int K) {
    int idx = blockIdx.x * blockDim.x + threadIdx.x;
    if (idx >= Npad * K) return;
    int n = idx / K;
    int k = idx - n * K;
    float v = (n < Nvalid) ? src[(size_t)(k + kOff) * ld + n] : 0.0f;
    dst[idx] = f2bf(v);
}

__global__ void k_f32_to_bf16(unsigned short* __restrict__ dst,
                              const float* __restrict__ src, int n) {
    int i = blockIdx.x * blockDim.x + threadIdx.x;
    if (i < n) dst[i] = f2bf(src[i]);
}

__global__ void k_zero_f32(float* __restrict__ p, int n) {
    int i = blockIdx.x * blockDim.x + threadIdx.x;
    if (i < n) p[i] = 0.0f;
}

// ---------------------------------------------------------------------------
// U = x @ W1_top + b1 ; V = x @ W1_bot     (one wave per 16x16 output tile)
// A: xbf [800][512] bf16 row-major.  Wt: [512][512] bf16 (pre-transposed).
// ---------------------------------------------------------------------------
__global__ __launch_bounds__(32)
void k_uv_gemm(const unsigned short* __restrict__ xbf,
               const unsigned short* __restrict__ w1a,
               const unsigned short* __restrict__ w1b,
               const float* __restrict__ gb1,
               float* __restrict__ U, float* __restrict__ V) {
    const int nt = blockIdx.x;            // 0..31
    const int mt = blockIdx.y;            // 0..49
    const int which = blockIdx.z;         // 0 -> U, 1 -> V
    const unsigned short* wt = which ? w1b : w1a;
    float* out = which ? V : U;

    const int l  = threadIdx.x;
    const int m  = l & 15;
    const int hi = l >> 4;
    const int n    = nt * 16 + m;
    const int arow = mt * 16 + m;

    v8f acc = VZERO8;
    for (int k0 = 0; k0 < HH; k0 += 32) {
        Frag a, b;
        a.i4[0] = *reinterpret_cast<const v4i*>(xbf + arow * 512 + k0 + hi * 8);
        a.i4[1] = *reinterpret_cast<const v4i*>(xbf + arow * 512 + k0 + 16 + hi * 8);
        b.i4[0] = *reinterpret_cast<const v4i*>(wt + n * 512 + k0 + hi * 8);
        b.i4[1] = *reinterpret_cast<const v4i*>(wt + n * 512 + k0 + 16 + hi * 8);
        acc = wmma_bf16(a.bf, b.bf, acc);
    }
    const float bb = which ? 0.0f : gb1[n];
#pragma unroll
    for (int r = 0; r < 8; ++r) {
        const int row = mt * 16 + r + hi * 8;
        out[(size_t)row * HH + n] = acc[r] + bb;
    }
}

// ---------------------------------------------------------------------------
// Fused relation kernel: per (64-pair tile, batch b), 512 threads = 16 waves.
//   h1 = relu(U[i] + V[j])            -> LDS bf16 [64][512]
//   h2 = relu(h1 @ W2 + b2)           -> LDS bf16 [64][512]   (WMMA)
//   h3 = relu(h2 @ W3 + b3)                                    (WMMA)
//   agg[b,:] += sum over valid rows (p<625 && i!=j)            (f32 atomics)
// Wave w: M-tile = w>>2, N-quadrant = w&3 (8 N-tiles of 16 cols each).
// M=64 blocking amortizes the per-WG read of W2+W3 (1 MB) 4x vs M=16,
// bringing L2 weight traffic (~320 MB total) under the matrix-op roofline.
// Dynamic LDS: 128 KB (CDNA5 allows up to 320 KB per workgroup).
// ---------------------------------------------------------------------------
__global__ __launch_bounds__(512)
void k_relation_fused(const float* __restrict__ U, const float* __restrict__ V,
                      const unsigned short* __restrict__ w2t,
                      const unsigned short* __restrict__ w3t,
                      const float* __restrict__ b2, const float* __restrict__ b3,
                      float* __restrict__ agg) {
    extern __shared__ unsigned short smem[];
    unsigned short* h1 = smem;                     // [64][512] bf16
    unsigned short* h2 = smem + MTILE * HH;        // [64][512] bf16

    const int tile = blockIdx.x;   // 0..9
    const int b    = blockIdx.y;   // 0..31
    const int t    = threadIdx.x;

    // ---- stage 1: h1 = relu(U_i + V_j), padded rows -> 0 ----
    for (int e = t; e < MTILE * HH; e += 512) {
        const int r = e >> 9;
        const int k = e & 511;
        const int p = tile * MTILE + r;
        float val = 0.0f;
        if (p < PAIRS) {
            const int i = p / NN;
            const int j = p - i * NN;
            val = U[(size_t)(b * NN + i) * HH + k] + V[(size_t)(b * NN + j) * HH + k];
            val = fmaxf(val, 0.0f);
        }
        h1[e] = f2bf(val);
    }
    __syncthreads();

    const int w  = t >> 5;     // 0..15
    const int l  = t & 31;
    const int m  = l & 15;
    const int hi = l >> 4;
    const int mt = w >> 2;     // M-tile 0..3
    const int nq = w & 3;      // N-quadrant 0..3
    const int abase = (mt * 16 + m) * 512;
    const v8f vz = VZERO8;
    v8f acc[8];

    // ---- layer 2: h2 = relu(h1 @ W2 + b2) ----
#pragma unroll
    for (int s = 0; s < 8; ++s) acc[s] = vz;
    for (int k0 = 0; k0 < HH; k0 += 32) {
        Frag a;
        a.i4[0] = *reinterpret_cast<const v4i*>(h1 + abase + k0 + hi * 8);
        a.i4[1] = *reinterpret_cast<const v4i*>(h1 + abase + k0 + 16 + hi * 8);
#pragma unroll
        for (int s = 0; s < 8; ++s) {
            const int n = (nq * 8 + s) * 16 + m;
            Frag bf;
            bf.i4[0] = *reinterpret_cast<const v4i*>(w2t + n * 512 + k0 + hi * 8);
            bf.i4[1] = *reinterpret_cast<const v4i*>(w2t + n * 512 + k0 + 16 + hi * 8);
            acc[s] = wmma_bf16(a.bf, bf.bf, acc[s]);
        }
    }
#pragma unroll
    for (int s = 0; s < 8; ++s) {
        const int n = (nq * 8 + s) * 16 + m;
        const float bb = b2[n];
#pragma unroll
        for (int r = 0; r < 8; ++r) {
            const int row = mt * 16 + r + hi * 8;
            h2[row * 512 + n] = f2bf(fmaxf(acc[s][r] + bb, 0.0f));
        }
    }
    __syncthreads();

    // ---- layer 3: h3 = relu(h2 @ W3 + b3), masked pair-sum into agg ----
#pragma unroll
    for (int s = 0; s < 8; ++s) acc[s] = vz;
    for (int k0 = 0; k0 < HH; k0 += 32) {
        Frag a;
        a.i4[0] = *reinterpret_cast<const v4i*>(h2 + abase + k0 + hi * 8);
        a.i4[1] = *reinterpret_cast<const v4i*>(h2 + abase + k0 + 16 + hi * 8);
#pragma unroll
        for (int s = 0; s < 8; ++s) {
            const int n = (nq * 8 + s) * 16 + m;
            Frag bf;
            bf.i4[0] = *reinterpret_cast<const v4i*>(w3t + n * 512 + k0 + hi * 8);
            bf.i4[1] = *reinterpret_cast<const v4i*>(w3t + n * 512 + k0 + 16 + hi * 8);
            acc[s] = wmma_bf16(a.bf, bf.bf, acc[s]);
        }
    }
#pragma unroll
    for (int s = 0; s < 8; ++s) {
        const int n = (nq * 8 + s) * 16 + m;
        const float bb = b3[n];
        float sum = 0.0f;
#pragma unroll
        for (int r = 0; r < 8; ++r) {
            const int row = mt * 16 + r + hi * 8;
            const int p = tile * MTILE + row;
            const int i = p / NN;
            const int j = p - i * NN;
            const float val = fmaxf(acc[s][r] + bb, 0.0f);
            if (p < PAIRS && i != j) sum += val;
        }
        atomicAdd(&agg[(size_t)b * HH + n], sum);
    }
}

// ---------------------------------------------------------------------------
// Head GEMM: M=32 rows, K=512. One wave per 16x16 tile.
// out_bf != null -> bf16 store (with relu); else f32 store to logits.
// ---------------------------------------------------------------------------
__global__ __launch_bounds__(32)
void k_head_gemm(const unsigned short* __restrict__ A,
                 const unsigned short* __restrict__ wt,
                 const float* __restrict__ bias,
                 unsigned short* __restrict__ out_bf,
                 float* __restrict__ out_f32,
                 int Nvalid, int Nstore, int do_relu) {
    const int nt = blockIdx.x;
    const int mt = blockIdx.y;           // 0..1 (M = 32)
    const int l  = threadIdx.x;
    const int m  = l & 15;
    const int hi = l >> 4;
    const int n    = nt * 16 + m;
    const int arow = mt * 16 + m;

    v8f acc = VZERO8;
    for (int k0 = 0; k0 < HH; k0 += 32) {
        Frag a, b;
        a.i4[0] = *reinterpret_cast<const v4i*>(A + arow * 512 + k0 + hi * 8);
        a.i4[1] = *reinterpret_cast<const v4i*>(A + arow * 512 + k0 + 16 + hi * 8);
        b.i4[0] = *reinterpret_cast<const v4i*>(wt + n * 512 + k0 + hi * 8);
        b.i4[1] = *reinterpret_cast<const v4i*>(wt + n * 512 + k0 + 16 + hi * 8);
        acc = wmma_bf16(a.bf, b.bf, acc);
    }
    if (n < Nvalid) {
        const float bb = bias[n];
#pragma unroll
        for (int r = 0; r < 8; ++r) {
            const int row = mt * 16 + r + hi * 8;
            float v = acc[r] + bb;
            if (do_relu) v = fmaxf(v, 0.0f);
            if (out_bf) out_bf[(size_t)row * Nstore + n] = f2bf(v);
            else        out_f32[(size_t)row * Nstore + n] = v;
        }
    }
}

// ---------------------------------------------------------------------------
// Host-side launcher
// ---------------------------------------------------------------------------
extern "C" void kernel_launch(void* const* d_in, const int* in_sizes, int n_in,
                              void* d_out, int out_size, void* d_ws, size_t ws_size,
                              hipStream_t stream) {
    (void)in_sizes; (void)n_in; (void)out_size; (void)ws_size;

    const float* x    = (const float*)d_in[0];
    const float* g_w1 = (const float*)d_in[1];
    const float* g_b1 = (const float*)d_in[2];
    const float* g_w2 = (const float*)d_in[3];
    const float* g_b2 = (const float*)d_in[4];
    const float* g_w3 = (const float*)d_in[5];
    const float* g_b3 = (const float*)d_in[6];
    const float* f_w1 = (const float*)d_in[7];
    const float* f_b1 = (const float*)d_in[8];
    const float* f_w2 = (const float*)d_in[9];
    const float* f_b2 = (const float*)d_in[10];
    const float* f_w3 = (const float*)d_in[11];
    const float* f_b3 = (const float*)d_in[12];
    float* logits = (float*)d_out;

    char* ws = (char*)d_ws;
    size_t off = 0;
    auto take = [&](size_t bytes) -> char* {
        char* p = ws + off;
        off = (off + bytes + 255) & ~(size_t)255;
        return p;
    };

    unsigned short* W1A = (unsigned short*)take((size_t)512 * 512 * 2);
    unsigned short* W1B = (unsigned short*)take((size_t)512 * 512 * 2);
    unsigned short* W2T = (unsigned short*)take((size_t)512 * 512 * 2);
    unsigned short* W3T = (unsigned short*)take((size_t)512 * 512 * 2);
    unsigned short* F1T = (unsigned short*)take((size_t)512 * 512 * 2);
    unsigned short* F2T = (unsigned short*)take((size_t)512 * 512 * 2);
    unsigned short* F3T = (unsigned short*)take((size_t)CPAD * 512 * 2);
    unsigned short* XBF = (unsigned short*)take((size_t)ROWS * 512 * 2);
    unsigned short* AGB = (unsigned short*)take((size_t)32 * 512 * 2);
    unsigned short* Z1  = (unsigned short*)take((size_t)32 * 512 * 2);
    unsigned short* Z2  = (unsigned short*)take((size_t)32 * 512 * 2);
    float* U   = (float*)take((size_t)ROWS * 512 * 4);
    float* V   = (float*)take((size_t)ROWS * 512 * 4);
    float* AGG = (float*)take((size_t)32 * 512 * 4);

    const int TB = 256;
    // Weight prep (transpose + bf16)
    k_transpose_bf16<<<dim3((512 * 512 + TB - 1) / TB), TB, 0, stream>>>(W1A, g_w1,   0, 512,  512,  512, 512);
    k_transpose_bf16<<<dim3((512 * 512 + TB - 1) / TB), TB, 0, stream>>>(W1B, g_w1, 512, 512,  512,  512, 512);
    k_transpose_bf16<<<dim3((512 * 512 + TB - 1) / TB), TB, 0, stream>>>(W2T, g_w2,   0, 512,  512,  512, 512);
    k_transpose_bf16<<<dim3((512 * 512 + TB - 1) / TB), TB, 0, stream>>>(W3T, g_w3,   0, 512,  512,  512, 512);
    k_transpose_bf16<<<dim3((512 * 512 + TB - 1) / TB), TB, 0, stream>>>(F1T, f_w1,   0, 512,  512,  512, 512);
    k_transpose_bf16<<<dim3((512 * 512 + TB - 1) / TB), TB, 0, stream>>>(F2T, f_w2,   0, 512,  512,  512, 512);
    k_transpose_bf16<<<dim3((CPAD * 512 + TB - 1) / TB), TB, 0, stream>>>(F3T, f_w3,  0, CC, CPAD,  CC, 512);

    k_f32_to_bf16<<<dim3((ROWS * 512 + TB - 1) / TB), TB, 0, stream>>>(XBF, x, ROWS * 512);
    k_zero_f32<<<dim3((32 * 512 + TB - 1) / TB), TB, 0, stream>>>(AGG, 32 * 512);

    // U/V factorized layer-1 GEMMs
    k_uv_gemm<<<dim3(32, ROWS / 16, 2), 32, 0, stream>>>(XBF, W1A, W1B, g_b1, U, V);

    // Fused relation core: 64 pairs/WG, 16 waves, 128 KB dynamic LDS
    const size_t fused_lds = (size_t)2 * MTILE * HH * sizeof(unsigned short);
    k_relation_fused<<<dim3(PT64, BB), 512, fused_lds, stream>>>(U, V, W2T, W3T, g_b2, g_b3, AGG);

    // Head
    k_f32_to_bf16<<<dim3((32 * 512 + TB - 1) / TB), TB, 0, stream>>>(AGB, AGG, 32 * 512);
    k_head_gemm<<<dim3(512 / 16, 2), 32, 0, stream>>>(AGB, F1T, f_b1, Z1, nullptr, 512, 512, 1);
    k_head_gemm<<<dim3(512 / 16, 2), 32, 0, stream>>>(Z1, F2T, f_b2, Z2, nullptr, 512, 512, 1);
    k_head_gemm<<<dim3(CPAD / 16, 2), 32, 0, stream>>>(Z2, F3T, f_b3, nullptr, logits, CC, CC, 0);
}
